// RelativeMultiheadAttention_60241211294422
// MI455X (gfx1250) — matrix operand order repaired and checked by reading, not compiled, not run on hardware
//
#include <hip/hip_runtime.h>

#define EMBED 1024
#define HEADS 16
#define HDIM  64
#define QLEN  512
#define KLEN  1024
#define BATCH 8
#define NEGBIG -1e20f
#define SCP 1032   // LDS pitch (floats) for score strip rows
#define PRP 1032   // LDS pitch (floats) for position strip rows

typedef __attribute__((ext_vector_type(16))) __bf16         v16bf;
typedef __attribute__((ext_vector_type(8)))  float          v8f;
typedef __attribute__((ext_vector_type(8)))  unsigned short v8us;

union FragBF { v16bf v; v8us h[2]; unsigned short s[16]; };
union AccF   { v8f v; float f[8]; };

__device__ inline unsigned short f2bf(float x){
  union { float f; unsigned u; } c; c.f = x;
  unsigned r = c.u + 0x7FFFu + ((c.u >> 16) & 1u);   // round-to-nearest-even
  return (unsigned short)(r >> 16);
}

// A fragment (16x32 bf16, row-major source, stride in elements).
// Lanes 0-15: row M=lane, K chunks {0..7} and {16..23}; lanes 16-31: {8..15},{24..31}.
__device__ inline v16bf ldFragA(const unsigned short* base, int rowBase, int rowMax,
                                int stride, int kBase){
  int lane = threadIdx.x & 31;
  int row  = rowBase + (lane & 15);
  if (row > rowMax) row = rowMax;
  int k0   = kBase + ((lane >> 4) << 3);
  const unsigned short* p = base + (size_t)row * stride + k0;
  FragBF f;
  f.h[0] = *(const v8us*)(p);
  f.h[1] = *(const v8us*)(p + 16);
  return f.v;
}

// B fragment (32x16 bf16); memory holds B^T as [n][k] row-major.
// Lanes 0-15: col N=lane, K chunks {0..7} and {16..23}; lanes 16-31: {8..15},{24..31}.
__device__ inline v16bf ldFragB(const unsigned short* base, int colBase, int colMax,
                                int stride, int kBase){
  int lane = threadIdx.x & 31;
  int col  = colBase + (lane & 15);
  if (col > colMax) col = colMax;
  int k0   = kBase + ((lane >> 4) << 3);
  const unsigned short* p = base + (size_t)col * stride + k0;
  FragBF f;
  f.h[0] = *(const v8us*)(p);
  f.h[1] = *(const v8us*)(p + 16);
  return f.v;
}

__device__ inline v8f wmma_bf16(v16bf a, v16bf b, v8f c){
  return __builtin_amdgcn_wmma_f32_16x16x32_bf16(false, a, false, b, (short)0, c,
                                                 false, false);
}

// ---------------- prep kernels ----------------

__global__ void cvt_bf16_kernel(const float* __restrict__ in,
                                unsigned short* __restrict__ out, int n){
  int i = blockIdx.x * blockDim.x + threadIdx.x;
  if (i < n) out[i] = f2bf(in[i]);
}

// Wt[n][k] = bf16(W[k][n]); W is 1024x1024 fp32 row-major.
__global__ void transposeW_kernel(const float* __restrict__ W,
                                  unsigned short* __restrict__ Wt){
  __shared__ float t[32][33];
  int bx = blockIdx.x * 32, by = blockIdx.y * 32;
  for (int j = threadIdx.y; j < 32; j += 8)
    t[j][threadIdx.x] = W[(size_t)(by + j) * EMBED + bx + threadIdx.x];
  __syncthreads();
  for (int j = threadIdx.y; j < 32; j += 8)
    Wt[(size_t)(bx + j) * EMBED + by + threadIdx.x] = f2bf(t[threadIdx.x][j]);
}

// ---------------- generic bf16 WMMA GEMM: C = A(MxK) * BT^T + bias ----------------
// grid = (EMBED/64, M/16); block = 32 (one wave). Each wave: 16x64 output strip.
// Epilogue modes:
//  0: Q proj  -> qU/qV [b][h][q][d] bf16 (adds U/V)
//  1: K proj  -> kpr [b][h][k][d] bf16
//  2: V proj  -> vT  [b][h][d][k] bf16 (transposed)
//  3: R proj  -> Rpr [h][k'][d] bf16
//  4: out proj-> fp32 d_out (row-major M x 1024)
__global__ void gemm_bf16_kernel(const unsigned short* __restrict__ A,
                                 const unsigned short* __restrict__ BT,
                                 const float* __restrict__ bias,
                                 int M, int mode,
                                 unsigned short* __restrict__ out0,
                                 unsigned short* __restrict__ out1,
                                 float* __restrict__ outF,
                                 const float* __restrict__ U,
                                 const float* __restrict__ V){
  int n0 = blockIdx.x * 64;
  int m0 = blockIdx.y * 16;
  AccF c[4];
  #pragma unroll
  for (int j = 0; j < 4; j++)
    #pragma unroll
    for (int r = 0; r < 8; r++) c[j].f[r] = 0.0f;

  for (int ks = 0; ks < EMBED; ks += 32){
    v16bf a = ldFragA(A, m0, M - 1, EMBED, ks);
    #pragma unroll
    for (int j = 0; j < 4; j++){
      v16bf b = ldFragB(BT, n0 + j * 16, EMBED - 1, EMBED, ks);
      c[j].v = wmma_bf16(a, b, c[j].v);
    }
  }

  int lane = threadIdx.x & 31;
  int nn = lane & 15;
  int mB = (lane >> 4) << 3;
  for (int j = 0; j < 4; j++){
    int col = n0 + j * 16 + nn;
    float bv = bias ? bias[col] : 0.0f;
    int h = col >> 6, d = col & 63;
    #pragma unroll
    for (int r = 0; r < 8; r++){
      int m = m0 + mB + r;
      float val = c[j].f[r] + bv;
      if (mode == 0){
        int ql = m >> 3, b = m & 7;
        size_t idx = ((((size_t)b * HEADS + h) * QLEN) + ql) * HDIM + d;
        out0[idx] = f2bf(val + U[h * HDIM + d]);
        out1[idx] = f2bf(val + V[h * HDIM + d]);
      } else if (mode == 1){
        int kq = m >> 3, b = m & 7;
        size_t idx = ((((size_t)b * HEADS + h) * KLEN) + kq) * HDIM + d;
        out0[idx] = f2bf(val);
      } else if (mode == 2){
        int kq = m >> 3, b = m & 7;
        size_t idx = ((((size_t)b * HEADS + h) * HDIM) + d) * KLEN + kq;
        out0[idx] = f2bf(val);
      } else if (mode == 3){
        size_t idx = (((size_t)h * KLEN) + m) * HDIM + d;
        out0[idx] = f2bf(val);
      } else {
        outF[(size_t)m * EMBED + col] = val;
      }
    }
  }
}

// ---------------- fused attention ----------------
// grid = (QLEN/32, BATCH*HEADS); block = 256 (8 waves).
// LDS: sc[32][SCP] score strip + pr[33][PRP] raw-position strip (reused as reduce scratch).
__global__ void __launch_bounds__(256, 1)
attn_fused_kernel(const unsigned short* __restrict__ qU,
                  const unsigned short* __restrict__ qV,
                  const unsigned short* __restrict__ kpr,
                  const unsigned short* __restrict__ vT,
                  const unsigned short* __restrict__ Rpr,
                  const unsigned char*  __restrict__ maskb,
                  const float*          __restrict__ pad,
                  unsigned short*       __restrict__ alpha){
  extern __shared__ float smem[];
  float* sc = smem;                 // 32 * SCP
  float* pr = smem + 32 * SCP;      // 33 * PRP

  int q0 = blockIdx.x * 32;
  int bh = blockIdx.y;
  int b = bh >> 4, h = bh & 15;
  int wave = threadIdx.x >> 5;
  int lane = threadIdx.x & 31;
  int nn = lane & 15;
  int mB = (lane >> 4) << 3;

  const unsigned short* qUbh = qU  + (((size_t)b * HEADS + h) * QLEN) * HDIM;
  const unsigned short* qVbh = qV  + (((size_t)b * HEADS + h) * QLEN) * HDIM;
  const unsigned short* kbh  = kpr + (((size_t)b * HEADS + h) * KLEN) * HDIM;
  const unsigned short* vbh  = vT  + (((size_t)b * HEADS + h) * HDIM) * KLEN;
  const unsigned short* Rh   = Rpr + ((size_t)h * KLEN) * HDIM;

  // Phase A: Praw strip rows q0..q0+32 (3 mtiles of 16; keep local rows < 33).
  for (int task = wave; task < 192; task += 8){
    int mt = task >> 6;      // 0..2
    int nt = task & 63;      // 0..63
    AccF c;
    #pragma unroll
    for (int r = 0; r < 8; r++) c.f[r] = 0.0f;
    #pragma unroll
    for (int ks = 0; ks < HDIM; ks += 32){
      v16bf a  = ldFragA(qVbh, q0 + mt * 16, QLEN - 1, HDIM, ks);
      v16bf br = ldFragB(Rh, nt * 16, KLEN - 1, HDIM, ks);
      c.v = wmma_bf16(a, br, c.v);
    }
    #pragma unroll
    for (int r = 0; r < 8; r++){
      int ml = mt * 16 + mB + r;
      if (ml < 33) pr[ml * PRP + nt * 16 + nn] = c.f[r];
    }
  }
  __syncthreads();

  // Phase B: content GEMM + rel-shift gather + masks + scale -> sc.
  for (int task = wave; task < 128; task += 8){
    int mt = task >> 6;      // 0..1
    int nt = task & 63;      // 0..63
    AccF c;
    #pragma unroll
    for (int r = 0; r < 8; r++) c.f[r] = 0.0f;
    #pragma unroll
    for (int ks = 0; ks < HDIM; ks += 32){
      v16bf a  = ldFragA(qUbh, q0 + mt * 16, QLEN - 1, HDIM, ks);
      v16bf bk = ldFragB(kbh, nt * 16, KLEN - 1, HDIM, ks);
      c.v = wmma_bf16(a, bk, c.v);
    }
    #pragma unroll
    for (int r = 0; r < 8; r++){
      int ml = mt * 16 + mB + r;          // local q row 0..31
      int q = q0 + ml;
      int k = nt * 16 + nn;
      int diff = k - q;
      float pv;
      if (diff <= 512)      pv = pr[ml * PRP + (k + 511 - q)];
      else if (diff == 513) pv = 0.0f;
      else                  pv = pr[(ml + 1) * PRP + (k - q - 514)];
      float s = (c.f[r] + pv) * 0.03125f;      // 1/sqrt(1024)
      if (maskb[(size_t)q * KLEN + k]) s = NEGBIG;
      if (k >= 512 && (pad[b * QLEN + q] == 0.0f || pad[b * QLEN + (k - 512)] == 0.0f))
        s = NEGBIG;
      sc[ml * SCP + k] = s;
    }
  }
  __syncthreads();

  // Phase C: softmax over k (1024) per row; 8 threads/row, 128 elems each.
  {
    int row = threadIdx.x >> 3;
    int seg = threadIdx.x & 7;
    float* rp = sc + row * SCP + seg * 128;
    float mx = -3.0e38f;
    for (int i = 0; i < 128; i++) mx = fmaxf(mx, rp[i]);
    pr[row * 8 + seg] = mx;
    __syncthreads();
    float rm = pr[row * 8];
    #pragma unroll
    for (int i = 1; i < 8; i++) rm = fmaxf(rm, pr[row * 8 + i]);
    float sum = 0.0f;
    for (int i = 0; i < 128; i++){ float e = __expf(rp[i] - rm); rp[i] = e; sum += e; }
    __syncthreads();
    pr[row * 8 + seg] = sum;
    __syncthreads();
    float rs = 0.0f;
    #pragma unroll
    for (int i = 0; i < 8; i++) rs += pr[row * 8 + i];
    float inv = 1.0f / rs;
    for (int i = 0; i < 128; i++) rp[i] *= inv;
  }
  __syncthreads();

  // Phase D: attn x v ; one 16x16 tile per wave (mt 0..1, nt 0..3), K=1024.
  {
    int mt = wave >> 2;
    int nt = wave & 3;
    AccF c;
    #pragma unroll
    for (int r = 0; r < 8; r++) c.f[r] = 0.0f;
    for (int ks = 0; ks < KLEN; ks += 32){
      int rowl = mt * 16 + (lane & 15);
      int k0 = ks + ((lane >> 4) << 3);
      const float* p = sc + rowl * SCP + k0;
      FragBF a;
      #pragma unroll
      for (int i = 0; i < 8; i++) a.s[i]     = f2bf(p[i]);
      #pragma unroll
      for (int i = 0; i < 8; i++) a.s[8 + i] = f2bf(p[16 + i]);
      v16bf bv = ldFragB(vbh, nt * 16, HDIM - 1, KLEN, ks);
      c.v = wmma_bf16(a.v, bv, c.v);
    }
    #pragma unroll
    for (int r = 0; r < 8; r++){
      int q = q0 + mt * 16 + mB + r;
      int d = nt * 16 + nn;
      alpha[((size_t)q * BATCH + b) * EMBED + h * HDIM + d] = f2bf(c.f[r]);
    }
  }
}

// ---------------- host launcher ----------------

extern "C" void kernel_launch(void* const* d_in, const int* in_sizes, int n_in,
                              void* d_out, int out_size, void* d_ws, size_t ws_size,
                              hipStream_t stream){
  (void)in_sizes; (void)n_in; (void)out_size; (void)ws_size;
  const float* query = (const float*)d_in[0];
  const float* key   = (const float*)d_in[1];
  const float* value = (const float*)d_in[2];
  const float* pos   = (const float*)d_in[3];
  const float* U     = (const float*)d_in[4];
  const float* V     = (const float*)d_in[5];
  const unsigned char* maskb = (const unsigned char*)d_in[6];  // jax bool = 1 byte
  const float* pad   = (const float*)d_in[7];
  const float* Wq = (const float*)d_in[8];  const float* bq = (const float*)d_in[9];
  const float* Wk = (const float*)d_in[10]; const float* bk = (const float*)d_in[11];
  const float* Wv = (const float*)d_in[12]; const float* bv = (const float*)d_in[13];
  const float* Wpos = (const float*)d_in[14];
  const float* Wo = (const float*)d_in[15]; const float* bo = (const float*)d_in[16];
  float* out = (float*)d_out;

  char* ws = (char*)d_ws;
  size_t off = 0;
  auto alloc = [&](size_t bytes){
    size_t o = off; off += (bytes + 255) & ~(size_t)255; return o;
  };
  const size_t WB = (size_t)EMBED * EMBED * 2;      // bf16 weight
  unsigned short* WqT = (unsigned short*)(ws + alloc(WB));
  unsigned short* WkT = (unsigned short*)(ws + alloc(WB));
  unsigned short* WvT = (unsigned short*)(ws + alloc(WB));
  unsigned short* WpT = (unsigned short*)(ws + alloc(WB));
  unsigned short* WoT = (unsigned short*)(ws + alloc(WB));
  unsigned short* qryB = (unsigned short*)(ws + alloc((size_t)QLEN * BATCH * EMBED * 2));
  unsigned short* keyB = (unsigned short*)(ws + alloc((size_t)KLEN * BATCH * EMBED * 2));
  unsigned short* valB = (unsigned short*)(ws + alloc((size_t)KLEN * BATCH * EMBED * 2));
  unsigned short* posB = (unsigned short*)(ws + alloc((size_t)KLEN * EMBED * 2));
  unsigned short* qUb = (unsigned short*)(ws + alloc((size_t)BATCH * HEADS * QLEN * HDIM * 2));
  unsigned short* qVb = (unsigned short*)(ws + alloc((size_t)BATCH * HEADS * QLEN * HDIM * 2));
  unsigned short* kprb = (unsigned short*)(ws + alloc((size_t)BATCH * HEADS * KLEN * HDIM * 2));
  unsigned short* vTb  = (unsigned short*)(ws + alloc((size_t)BATCH * HEADS * HDIM * KLEN * 2));
  unsigned short* Rprb = (unsigned short*)(ws + alloc((size_t)HEADS * KLEN * HDIM * 2));
  unsigned short* alphab = (unsigned short*)(ws + alloc((size_t)QLEN * BATCH * EMBED * 2));

  const int SMEM_BYTES = (32 * SCP + 33 * PRP) * 4;   // ~262 KB of the 320 KB WGP LDS
  (void)hipFuncSetAttribute((const void*)attn_fused_kernel,
                            hipFuncAttributeMaxDynamicSharedMemorySize, SMEM_BYTES);

  dim3 tb(32, 8);
  dim3 tg(EMBED / 32, EMBED / 32);
  transposeW_kernel<<<tg, tb, 0, stream>>>(Wq, WqT);
  transposeW_kernel<<<tg, tb, 0, stream>>>(Wk, WkT);
  transposeW_kernel<<<tg, tb, 0, stream>>>(Wv, WvT);
  transposeW_kernel<<<tg, tb, 0, stream>>>(Wpos, WpT);
  transposeW_kernel<<<tg, tb, 0, stream>>>(Wo, WoT);

  int nQ = QLEN * BATCH * EMBED, nK = KLEN * BATCH * EMBED, nP = KLEN * EMBED;
  cvt_bf16_kernel<<<(nQ + 255) / 256, 256, 0, stream>>>(query, qryB, nQ);
  cvt_bf16_kernel<<<(nK + 255) / 256, 256, 0, stream>>>(key,   keyB, nK);
  cvt_bf16_kernel<<<(nK + 255) / 256, 256, 0, stream>>>(value, valB, nK);
  cvt_bf16_kernel<<<(nP + 255) / 256, 256, 0, stream>>>(pos,   posB, nP);

  // projections
  gemm_bf16_kernel<<<dim3(EMBED / 64, (QLEN * BATCH) / 16), 32, 0, stream>>>(
      qryB, WqT, bq, QLEN * BATCH, 0, qUb, qVb, nullptr, U, V);
  gemm_bf16_kernel<<<dim3(EMBED / 64, (KLEN * BATCH) / 16), 32, 0, stream>>>(
      keyB, WkT, bk, KLEN * BATCH, 1, kprb, nullptr, nullptr, nullptr, nullptr);
  gemm_bf16_kernel<<<dim3(EMBED / 64, (KLEN * BATCH) / 16), 32, 0, stream>>>(
      valB, WvT, bv, KLEN * BATCH, 2, vTb, nullptr, nullptr, nullptr, nullptr);
  gemm_bf16_kernel<<<dim3(EMBED / 64, KLEN / 16), 32, 0, stream>>>(
      posB, WpT, nullptr, KLEN, 3, Rprb, nullptr, nullptr, nullptr, nullptr);

  // fused relative attention
  attn_fused_kernel<<<dim3(QLEN / 32, BATCH * HEADS), 256, SMEM_BYTES, stream>>>(
      qUb, qVb, kprb, vTb, Rprb, maskb, pad, alphab);

  // output projection -> fp32 d_out
  gemm_bf16_kernel<<<dim3(EMBED / 64, (QLEN * BATCH) / 16), 32, 0, stream>>>(
      alphab, WoT, bo, QLEN * BATCH, 4, nullptr, nullptr, out, nullptr, nullptr);
}
